// bandAttn_Meta_upsample_81930796138535
// MI455X (gfx1250) — matrix operand order repaired and checked by reading, not compiled, not run on hardware
//
#include <hip/hip_runtime.h>
#include <math.h>

typedef _Float16 v16h __attribute__((ext_vector_type(16)));
typedef float    v8f  __attribute__((ext_vector_type(8)));

#define DEV_INLINE __device__ __forceinline__

DEV_INLINE float sigmoidf_(float v) { return 1.0f / (1.0f + expf(-v)); }
DEV_INLINE float reluf_(float v)    { return fmaxf(v, 0.0f); }

// ---------------------------------------------------------------------------
// Kernel 1: channel mean of x over (b, h, w).  x: [4][64][64][64]
// ---------------------------------------------------------------------------
__global__ __launch_bounds__(256) void chan_mean_k(const float* __restrict__ x,
                                                   float* __restrict__ cm) {
  __shared__ float red[256];
  const int c = blockIdx.x;
  const int t = threadIdx.x;
  float s = 0.0f;
  for (int b = 0; b < 4; ++b) {
    const float* base = x + (size_t)(b * 64 + c) * 4096;
    for (int i = t; i < 4096; i += 256) s += base[i];
  }
  red[t] = s;
  __syncthreads();
  for (int off = 128; off > 0; off >>= 1) {
    if (t < off) red[t] += red[t + off];
    __syncthreads();
  }
  if (t == 0) cm[c] = red[0] * (1.0f / 16384.0f);
}

// ---------------------------------------------------------------------------
// Kernel 2: the whole meta network on the 4x4 phase grid (single block).
// Outputs:
//   offXY[p][2][64]  (f32)   per-phase per-channel sampling offsets
//   Mh[p][64][64]    (f16)   per-phase mixing matrix  Wem[p] @ Wcm[p]
// ---------------------------------------------------------------------------
__global__ __launch_bounds__(256) void phase_net_k(
    const float* __restrict__ wc,  const float* __restrict__ we,
    const float* __restrict__ Wb1, const float* __restrict__ bb1,
    const float* __restrict__ Wb2, const float* __restrict__ bb2,
    const float* __restrict__ Wr1, const float* __restrict__ br1,
    const float* __restrict__ Wr2, const float* __restrict__ br2,
    const float* __restrict__ Wq1, const float* __restrict__ bq1,
    const float* __restrict__ Wq2, const float* __restrict__ bq2,
    const float* __restrict__ Wq3, const float* __restrict__ bq3,
    const float* __restrict__ Wo1, const float* __restrict__ bo1,
    const float* __restrict__ Wo2, const float* __restrict__ bo2,
    const float* __restrict__ Wo3, const float* __restrict__ bo3,
    const float* __restrict__ chanMean,
    float* __restrict__ offXY, _Float16* __restrict__ Mh) {
  __shared__ float cmS[64];
  __shared__ float embS[64][64];   // [channel][f*16 + phase]
  __shared__ float rS[64];         // [f*16 + phase]
  __shared__ float rwS[4][16];     // [expert][phase]
  __shared__ float wcS[4][8][64];
  __shared__ float weS[4][64][8];

  const int t = threadIdx.x;
  if (t < 64) cmS[t] = chanMean[t];
  for (int i = t; i < 2048; i += 256) {
    (&wcS[0][0][0])[i] = wc[i];
    (&weS[0][0][0])[i] = we[i];
  }
  __syncthreads();

  // ---- Step A: emb columns (64 columns = 4 meta-features x 16 phases) -----
  if (t < 64) {
    const int f = t >> 4, p = t & 15;
    const int phh = p >> 2, pww = p & 3;
    float m;
    if      (f == 0) m = 1.0f;
    else if (f == 1) m = 0.25f;
    else if (f == 2) m = (phh + 0.5f) * 0.25f - 0.5f;   // coor_h(h%4)
    else             m = (pww + 0.5f) * 0.25f - 0.5f;   // coor_w(w%4)

    float h1[64];
    for (int o = 0; o < 64; ++o) {
      float acc = bb1[o];
      const float* row = Wb1 + o * 64;
      if (f == 0) { for (int cc = 0; cc < 64; ++cc) acc += row[cc] * (m * cmS[cc]); }
      else        { for (int cc = 0; cc < 64; ++cc) acc += row[cc] * m; }
      h1[o] = reluf_(acc);
    }
    for (int o = 0; o < 64; ++o) {
      float acc = bb2[o];
      const float* row = Wb2 + o * 64;
      for (int cc = 0; cc < 64; ++cc) acc += row[cc] * h1[cc];
      embS[o][t] = reluf_(acc);
    }
  }
  __syncthreads();

  // ---- Step B: routing_1  r[f,p] -----------------------------------------
  if (t < 64) {
    float h1[64];
    for (int o = 0; o < 64; ++o) {
      float acc = br1[o];
      const float* row = Wr1 + o * 64;
      for (int cc = 0; cc < 64; ++cc) acc += row[cc] * embS[cc][t];
      h1[o] = reluf_(acc);
    }
    float s2 = br2[0];
    for (int cc = 0; cc < 64; ++cc) s2 += Wr2[cc] * h1[cc];
    rS[t] = sigmoidf_(s2);
  }
  __syncthreads();

  // ---- Step C: routing_2  rw[e,p] ----------------------------------------
  if (t < 16) {
    float in4[4];
    for (int f = 0; f < 4; ++f) in4[f] = rS[f * 16 + t];
    float h1[64], h2[64];
    for (int o = 0; o < 64; ++o) {
      float acc = bq1[o];
      for (int f = 0; f < 4; ++f) acc += Wq1[o * 4 + f] * in4[f];
      h1[o] = reluf_(acc);
    }
    for (int o = 0; o < 64; ++o) {
      float acc = bq2[o];
      const float* row = Wq2 + o * 64;
      for (int cc = 0; cc < 64; ++cc) acc += row[cc] * h1[cc];
      h2[o] = reluf_(acc);
    }
    for (int e = 0; e < 4; ++e) {
      float acc = bq3[e];
      const float* row = Wq3 + e * 64;
      for (int cc = 0; cc < 64; ++cc) acc += row[cc] * h2[cc];
      rwS[e][t] = sigmoidf_(acc);
    }
  }
  __syncthreads();

  // ---- Step D: offset head, per (channel, phase) = 1024 combos -----------
  for (int it = 0; it < 4; ++it) {
    const int combo = t + it * 256;
    const int c = combo >> 4, p = combo & 15;
    float in4[4];
    for (int f = 0; f < 4; ++f) in4[f] = embS[c][f * 16 + p];
    float h1[64], h2[64];
    for (int o = 0; o < 64; ++o) {
      float acc = bo1[o];
      for (int f = 0; f < 4; ++f) acc += Wo1[o * 4 + f] * in4[f];
      h1[o] = reluf_(acc);
    }
    for (int o = 0; o < 64; ++o) {
      float acc = bo2[o];
      const float* row = Wo2 + o * 64;
      for (int cc = 0; cc < 64; ++cc) acc += row[cc] * h1[cc];
      h2[o] = reluf_(acc);
    }
    float ox = bo3[0], oy = bo3[1];
    for (int cc = 0; cc < 64; ++cc) { ox += Wo3[cc] * h2[cc]; oy += Wo3[64 + cc] * h2[cc]; }
    offXY[(p * 2 + 0) * 64 + c] = ox;
    offXY[(p * 2 + 1) * 64 + c] = oy;
  }
  __syncthreads();

  // ---- Step E: M[p] = (sum_e rw we_e) @ (sum_e rw wc_e),  f16 ------------
  for (int it = 0; it < 4; ++it) {
    const int idx = t + it * 256;   // (p, o) row
    const int p = idx >> 6, o = idx & 63;
    float wem[8];
    for (int kk = 0; kk < 8; ++kk) {
      float acc = 0.0f;
      for (int e = 0; e < 4; ++e) acc += rwS[e][p] * weS[e][o][kk];
      wem[kk] = acc;
    }
    for (int i = 0; i < 64; ++i) {
      float acc = 0.0f;
      for (int kk = 0; kk < 8; ++kk) {
        float wcm = 0.0f;
        for (int e = 0; e < 4; ++e) wcm += rwS[e][p] * wcS[e][kk][i];
        acc += wem[kk] * wcm;
      }
      Mh[((size_t)p * 64 + o) * 64 + i] = (_Float16)acc;
    }
  }
}

// ---------------------------------------------------------------------------
// Kernel 3: fused gather (deformable bilinear sample) + residual expert mix
// via WMMA f16.  out[b,co,H,W] = fea0 + M[phase] @ fea0.
//
// Key invariant exploited: for a fixed (channel, phase) the source coordinate
// steps by exactly 1.0 per phase-grid pixel, so the bilinear weights and the
// integer base indices are BLOCK CONSTANTS — computed once per block, the
// per-sample work is 2 int adds + 4 predicated coalesced loads + 5 FMAs.
//
// Grid: 4 batches x 16 phases x 8 chunks = 512 blocks, 256 threads (8 waves).
// ---------------------------------------------------------------------------
__global__ __launch_bounds__(256) void fused_main_k(
    const float* __restrict__ x, const float* __restrict__ offXY,
    const _Float16* __restrict__ Mh, float* __restrict__ out) {
  __shared__ _Float16 sM[64][64];    // mixing matrix for this phase (f16)
  __shared__ int   sIbx[64], sIby[64];  // per-channel integer sample bases
  __shared__ float sWx[64],  sWy[64];   // per-channel bilinear weights
  __shared__ float sF[8][64][17];       // per-wave gathered fea0 tile [ch][px]

  const int tid  = threadIdx.x;
  const int wave = tid >> 5;
  const int lane = tid & 31;

  const int bx    = blockIdx.x;
  const int chunk = bx & 7;
  const int q     = (bx >> 3) & 15;  // phase index
  const int b     = bx >> 7;
  const int ph    = q >> 2, pw = q & 3;

  // Stage M[phase] (8 KB) into LDS with 128-bit copies.
  {
    const float4* src = reinterpret_cast<const float4*>(Mh + (size_t)q * 4096);
    float4* dst = reinterpret_cast<float4*>(&sM[0][0]);
    dst[tid]       = src[tid];
    dst[tid + 256] = src[tid + 256];
  }
  // Per-(channel, phase) sampling constants: one floorf per channel per block.
  if (tid < 64) {
    const float ox = offXY[(q * 2 + 0) * 64 + tid];
    const float oy = offXY[(q * 2 + 1) * 64 + tid];
    const float ax = (pw + 0.5f) * 0.25f - 0.5f + ox;  // + j gives px(j)
    const float ay = (ph + 0.5f) * 0.25f - 0.5f + oy;  // + i gives py(i)
    const float fx = floorf(ax), fy = floorf(ay);
    sIbx[tid] = (int)fx;  sWx[tid] = ax - fx;
    sIby[tid] = (int)fy;  sWy[tid] = ay - fy;
  }
  __syncthreads();

  // A-fragments: per-phase constant, contiguous-k 16-bit A(16x32) layout, same k order as B.
  // lane L: row m = L%16; element e -> K = e + 16*(L/16).
  v16h afrag[4][2];
#pragma unroll
  for (int mt = 0; mt < 4; ++mt)
#pragma unroll
    for (int s = 0; s < 2; ++s)
#pragma unroll
      for (int e = 0; e < 16; ++e) {
        const int K = e + 16 * (lane >> 4);
        afrag[mt][s][e] = sM[16 * mt + (lane & 15)][32 * s + K];
      }

  const int   chalf = lane >> 4;
  const int   pxl   = lane & 15;
  const float* xb   = x + (size_t)b * 64 * 4096;
  __builtin_prefetch(xb, 0, 0);  // x[b] is 1 MB: L2-hot after first touch

  for (int iter = 0; iter < 4; ++iter) {
    const int pb = chunk * 512 + iter * 128 + wave * 16;  // tile base pixel
    const int i0 = pb >> 6;        // phase-grid row (same for all 16 pixels)
    const int j0 = pb & 63;        // phase-grid col base (16-aligned)
    const int hh = ph + 4 * i0;
    const int ww = pw + 4 * (j0 + pxl);

    // ---- gather: 64 channels x 16 pixels per wave; 32 samples per lane ----
    for (int t2 = 0; t2 < 32; ++t2) {
      const int c  = chalf * 32 + t2;
      const int xi = sIbx[c] + j0 + pxl;   // lane-consecutive -> coalesced
      const int yi = sIby[c] + i0;
      const float wx = sWx[c], wy = sWy[c];
      const float* img = xb + (size_t)c * 4096;

      const bool y0ok = (unsigned)yi < 64u;
      const bool y1ok = (unsigned)(yi + 1) < 64u;
      const bool x0ok = (unsigned)xi < 64u;
      const bool x1ok = (unsigned)(xi + 1) < 64u;
      const int  r0 = (yi < 0 ? 0 : (yi > 63 ? 63 : yi)) * 64;
      const int  r1 = (yi + 1 < 0 ? 0 : (yi + 1 > 63 ? 63 : yi + 1)) * 64;
      const int  c0 = xi < 0 ? 0 : (xi > 63 ? 63 : xi);
      const int  c1 = xi + 1 < 0 ? 0 : (xi + 1 > 63 ? 63 : xi + 1);

      const float v00 = (y0ok && x0ok) ? img[r0 + c0] : 0.0f;
      const float v01 = (y0ok && x1ok) ? img[r0 + c1] : 0.0f;
      const float v10 = (y1ok && x0ok) ? img[r1 + c0] : 0.0f;
      const float v11 = (y1ok && x1ok) ? img[r1 + c1] : 0.0f;

      const float top = v00 + wx * (v01 - v00);
      const float bot = v10 + wx * (v11 - v10);
      sF[wave][c][pxl] = top + wy * (bot - top);
    }
    __syncthreads();   // wave-local tiles; barrier keeps LDS ordering simple

    // ---- B fragments: 32x16 (KxN) f16, lane L: N = L%16, khalf = L/16 -----
    v16h bf0, bf1;
#pragma unroll
    for (int e = 0; e < 16; ++e) {
      bf0[e] = (_Float16)sF[wave][chalf * 16 + e][pxl];        // K = 0..31
      bf1[e] = (_Float16)sF[wave][32 + chalf * 16 + e][pxl];   // K = 32..63
    }

    // ---- 4 M-tiles x 2 K-steps; C seeded with the residual fea0 -----------
#pragma unroll
    for (int mt = 0; mt < 4; ++mt) {
      v8f acc;
#pragma unroll
      for (int v = 0; v < 8; ++v)
        acc[v] = sF[wave][16 * mt + v + 8 * chalf][pxl];   // residual preload

      acc = __builtin_amdgcn_wmma_f32_16x16x32_f16(
          false, afrag[mt][0], false, bf0, (short)0, acc, false, false);
      acc = __builtin_amdgcn_wmma_f32_16x16x32_f16(
          false, afrag[mt][1], false, bf1, (short)0, acc, false, false);

#pragma unroll
      for (int v = 0; v < 8; ++v) {
        const int co = 16 * mt + v + 8 * chalf;
        out[(((size_t)b * 64 + co) * 256 + hh) * 256 + ww] = acc[v];
      }
    }
    __syncthreads();
  }
}

// ---------------------------------------------------------------------------
// Launch
// ---------------------------------------------------------------------------
extern "C" void kernel_launch(void* const* d_in, const int* in_sizes, int n_in,
                              void* d_out, int out_size, void* d_ws, size_t ws_size,
                              hipStream_t stream) {
  (void)in_sizes; (void)n_in; (void)out_size; (void)ws_size;
  const float* x   = (const float*)d_in[0];
  const float* wc  = (const float*)d_in[1];
  const float* we  = (const float*)d_in[2];
  const float* Wb1 = (const float*)d_in[3];  const float* bb1 = (const float*)d_in[4];
  const float* Wb2 = (const float*)d_in[5];  const float* bb2 = (const float*)d_in[6];
  const float* Wr1 = (const float*)d_in[7];  const float* br1 = (const float*)d_in[8];
  const float* Wr2 = (const float*)d_in[9];  const float* br2 = (const float*)d_in[10];
  const float* Wq1 = (const float*)d_in[11]; const float* bq1 = (const float*)d_in[12];
  const float* Wq2 = (const float*)d_in[13]; const float* bq2 = (const float*)d_in[14];
  const float* Wq3 = (const float*)d_in[15]; const float* bq3 = (const float*)d_in[16];
  const float* Wo1 = (const float*)d_in[17]; const float* bo1 = (const float*)d_in[18];
  const float* Wo2 = (const float*)d_in[19]; const float* bo2 = (const float*)d_in[20];
  const float* Wo3 = (const float*)d_in[21]; const float* bo3 = (const float*)d_in[22];
  // d_in[23] = scale (always 4 for this configuration)

  float* ws        = (float*)d_ws;
  float* chanMean  = ws;              // 64 f32
  float* offXY     = ws + 64;         // 16*2*64 = 2048 f32
  _Float16* Mh     = (_Float16*)(ws + 64 + 2048);  // 16*64*64 f16 (16B-aligned)
  float* out       = (float*)d_out;

  chan_mean_k<<<64, 256, 0, stream>>>(x, chanMean);
  phase_net_k<<<1, 256, 0, stream>>>(wc, we, Wb1, bb1, Wb2, bb2,
                                     Wr1, br1, Wr2, br2,
                                     Wq1, bq1, Wq2, bq2, Wq3, bq3,
                                     Wo1, bo1, Wo2, bo2, Wo3, bo3,
                                     chanMean, offXY, Mh);
  fused_main_k<<<512, 256, 0, stream>>>(x, offXY, Mh, out);
}
